// FastCausalMultiheadAttention_18571438588266
// MI455X (gfx1250) — matrix operand order repaired and checked by reading, not compiled, not run on hardware
//
#include <hip/hip_runtime.h>
#include <hip/hip_bf16.h>
#include <math.h>

// ---------------- problem constants ----------------
#define TL 2048
#define BS 16
#define NH 8
#define DH 64
#define MF 266
#define MP 288            // 266 padded to 9*32 for K-loop of 32
#define BHT (BS * NH * TL)

#define DN 0.35355339059327373f     // 64^-0.25
#define QSCALE 0.125f               // 64^-0.5
#define RATIO 0.06131393394849658f  // 266^-0.5
#define EPSK 1.0e-4f
#define EPSD 1.0e-6f

typedef __attribute__((ext_vector_type(16))) __bf16 v16bf;
typedef __attribute__((ext_vector_type(8))) __bf16 v8bf;
typedef __attribute__((ext_vector_type(8))) float v8f;
typedef __attribute__((ext_vector_type(16))) float v16f;
typedef __attribute__((ext_vector_type(4))) unsigned v4u;
typedef __attribute__((ext_vector_type(8))) int v8i;
typedef __attribute__((ext_vector_type(4))) int v4i;

__device__ __forceinline__ v8f wmma_bf16(v16bf a, v16bf b, v8f c) {
  return __builtin_amdgcn_wmma_f32_16x16x32_bf16(false, a, false, b, (short)0, c,
                                                 false, false);
}

// A fragment (16x32 bf16), row-major LDS [rows x ld], origin (r0,k0).
// lanes 0-15: row=lane, K=0..15; lanes 16-31: K=16..31.
// ld multiple of 8 -> both halves are 16B-aligned ds_load_b128.
__device__ __forceinline__ v16bf frag_a(const __bf16* X, int ld, int r0, int k0,
                                        int lane) {
  const __bf16* p = X + (r0 + (lane & 15)) * ld + k0 + ((lane & 16) ? 16 : 0);
  v8bf lo = *(const v8bf*)p;         // K 0..7
  v8bf hi = *(const v8bf*)(p + 8);   // K 8..15
  return __builtin_shufflevector(lo, hi, 0, 1, 2, 3, 4, 5, 6, 7, 8, 9, 10, 11,
                                 12, 13, 14, 15);
}

// B fragment (32x16 bf16) from N-MAJOR LDS [N x ld] (K contiguous per row),
// origin (k0,n0). lanes 0-15: K=k0..k0+15; lanes 16-31: K=k0+16..k0+31.
// Two contiguous 16B ds_load_b128 per lane, no packing VALU.
__device__ __forceinline__ v16bf frag_bT(const __bf16* Bt, int ld, int k0,
                                         int n0, int lane) {
  const __bf16* p = Bt + (n0 + (lane & 15)) * ld + k0 + ((lane & 16) ? 16 : 0);
  v8bf lo = *(const v8bf*)p;
  v8bf hi = *(const v8bf*)(p + 8);
  return __builtin_shufflevector(lo, hi, 0, 1, 2, 3, 4, 5, 6, 7, 8, 9, 10, 11,
                                 12, 13, 14, 15);
}

// Same, fp32 N-major source (running state S^T), converted on load.
__device__ __forceinline__ v16bf frag_bT_f32(const float* Bt, int ld, int k0,
                                             int n0, int lane) {
  const float* p = Bt + (n0 + (lane & 15)) * ld + k0 + ((lane & 16) ? 16 : 0);
  v16f t = *(const v16f*)p;
  v16bf b;
#pragma unroll
  for (int i = 0; i < 16; ++i) b[i] = (__bf16)t[i];
  return b;
}

// ---- Tensor Data Mover: 1-D contiguous tile, 2-byte elements ----
// Builds a D# (group0/group1) per CDNA5 ISA and issues
// TENSOR_LOAD_TO_LDS. elems = number of 2-byte elements (<= 65535).
__device__ __forceinline__ void tdm_load_1d(unsigned lds_off, const void* gptr,
                                            unsigned elems) {
  unsigned long long ga = (unsigned long long)(size_t)gptr;
  v4u g0;
  g0[0] = 1u;                                      // count=1, user D#
  g0[1] = lds_off;                                 // lds_addr (bytes)
  g0[2] = (unsigned)(ga & 0xffffffffu);            // global_addr[31:0]
  g0[3] = (unsigned)((ga >> 32) & 0x1ffffffu) | 0x80000000u;  // [56:32], type=2
  v8i g1;
  g1[0] = (int)(1u << 16);                         // data_size=1 (2B), mask=0
  g1[1] = (int)((elems & 0xffffu) << 16);          // tensor_dim0[15:0]
  g1[2] = (int)(((elems >> 16) & 0xffffu) | (1u << 16));  // dim0 hi, dim1=1
  g1[3] = (int)((elems & 0xffffu) << 16);          // tile_dim0
  g1[4] = 0;                                       // tile_dim1/2 unused
  g1[5] = (int)elems;                              // tensor_dim0_stride lo
  g1[6] = 0;
  g1[7] = 0;
  v4i gz = {0, 0, 0, 0};
#if defined(__clang_major__) && (__clang_major__ >= 23)
  v8i gz8 = {0, 0, 0, 0, 0, 0, 0, 0};
  __builtin_amdgcn_tensor_load_to_lds(g0, g1, gz, gz, gz8, 0);
#else
  __builtin_amdgcn_tensor_load_to_lds(g0, g1, gz, gz, 0);
#endif
}

// monotonic-uint mapping for float atomicMax
__device__ __forceinline__ unsigned f2mono(float f) {
  int i = __float_as_int(f);
  return (i >= 0) ? ((unsigned)i | 0x80000000u) : ~(unsigned)i;
}
__device__ __forceinline__ float mono2f(unsigned u) {
  int i = (u & 0x80000000u) ? (int)(u ^ 0x80000000u) : (int)~u;
  return __int_as_float(i);
}

// ================= kernel 1: fused head projection =================
// y[row,:512] = x[row,:64] @ W^T + b (then *scale) -> [B,H,T,64] bf16
__global__ __launch_bounds__(256) void proj_kernel(
    const float* __restrict__ x, const float* __restrict__ W,
    const float* __restrict__ bias, __bf16* __restrict__ dst, float scale) {
  extern __shared__ __align__(16) char smem[];
  __bf16* Xs = (__bf16*)smem;                   // [64][64]
  __bf16* Ws = (__bf16*)(smem + 64 * 64 * 2);   // [512][64] N-major (=W layout)
  const int tid = threadIdx.x, lane = tid & 31, wave = tid >> 5;
  const int row0 = blockIdx.x * 64;

  for (int i = tid; i < 64 * 64; i += 256)
    Xs[i] = (__bf16)x[(size_t)row0 * 64 + i];
  for (int i = tid; i < 512 * 64; i += 256)
    Ws[i] = (__bf16)W[i];                       // direct: W is [o][d]
  __syncthreads();

  const int rt = wave & 3;    // 16-row tile
  const int chf = wave >> 2;  // 256-col half
  for (int ct = 0; ct < 16; ++ct) {
    const int n0 = chf * 256 + ct * 16;
    v8f acc = {};
#pragma unroll
    for (int k0 = 0; k0 < 64; k0 += 32) {
      v16bf a = frag_a(Xs, 64, rt * 16, k0, lane);
      v16bf b = frag_bT(Ws, 64, k0, n0, lane);
      acc = wmma_bf16(a, b, acc);
    }
    const int n = n0 + (lane & 15);
    const int h = n >> 6, d = n & 63;
    const int Mo = (lane >> 4) << 3;
#pragma unroll
    for (int v = 0; v < 8; ++v) {
      int row = row0 + rt * 16 + v + Mo;
      int t = row / BS, bi = row % BS;
      float val = (acc[v] + bias[n]) * scale;
      dst[(((size_t)(bi * NH + h)) * TL + t) * DH + d] = (__bf16)val;
    }
  }
}

// ================= kernel 2: Performer feature map =================
__global__ __launch_bounds__(256) void feature_kernel(
    const __bf16* __restrict__ src, const float* __restrict__ proj,
    __bf16* __restrict__ dst, unsigned* __restrict__ kmax, int is_query,
    int phase) {
  extern __shared__ __align__(16) char smem[];
  __bf16* Xs = (__bf16*)smem;                    // [64][64]
  __bf16* Pt = Xs + 64 * 64;                     // [MP][64] N-major (m rows)
  float* DD = (float*)((char*)(Pt + MP * 64));   // [64][MP]
  float* diag_s = DD + 64 * MP;                  // [64]
  float* mx_s = diag_s + 64;                     // [64]
  const int tid = threadIdx.x, lane = tid & 31, wave = tid >> 5;
  const int row0 = blockIdx.x * 64;

  if (wave == 0)  // X tile is a contiguous 8KB block: TDM it
    tdm_load_1d(0u, src + (size_t)row0 * 64, 64 * 64);
  for (int i = tid; i < MP * 64; i += 256) {
    int m = i >> 6, d = i & 63;
    Pt[i] = (m < MF) ? (__bf16)(proj[m * 64 + d] * DN) : (__bf16)0.0f;
  }
  if (wave == 0) __builtin_amdgcn_s_wait_tensorcnt(0);
  __syncthreads();

  for (int u = 0; u < 9; ++u) {  // 4x18 = 72 tiles, 9 per wave
    int ti = wave * 9 + u, rt = ti / 18, ct = ti % 18;
    v8f acc = {};
#pragma unroll
    for (int k0 = 0; k0 < 64; k0 += 32) {
      v16bf a = frag_a(Xs, 64, rt * 16, k0, lane);
      v16bf b = frag_bT(Pt, 64, k0, ct * 16, lane);
      acc = wmma_bf16(a, b, acc);
    }
    const int N = lane & 15, Mo = (lane >> 4) << 3;
#pragma unroll
    for (int v = 0; v < 8; ++v)
      DD[(rt * 16 + v + Mo) * MP + ct * 16 + N] = acc[v];
  }
  __syncthreads();

  if (tid < 64) {
    float dg = 0.f;
    for (int d = 0; d < 64; ++d) {
      float xv = (float)Xs[tid * 64 + d];
      dg += xv * xv;
    }
    diag_s[tid] = 0.5f * DN * DN * dg;
    float mx = -1.0e30f;
    for (int m = 0; m < MF; ++m) mx = fmaxf(mx, DD[tid * MP + m]);
    mx_s[tid] = mx;
    if (!is_query && phase == 0) atomicMax(kmax, f2mono(mx));
  }
  __syncthreads();

  if (!is_query && phase == 0) return;  // reduction pass only

  const float gmx = is_query ? 0.f : mono2f(*kmax);
  for (int i = tid; i < 64 * MP; i += 256) {
    int r = i / MP, m = i % MP;
    float val = 0.f;
    if (m < MF) {
      float mx = is_query ? mx_s[r] : gmx;
      val = RATIO * (__expf(DD[i] - diag_s[r] - mx) + EPSK);
    }
    dst[(size_t)(row0 + r) * MP + m] = (__bf16)val;
  }
}

// ================= kernel 3: chunked causal linear attention =================
// One workgroup per (b,h). State kept TRANSPOSED: St[d][m] fp32 in LDS.
__global__ __launch_bounds__(256) void attn_kernel(
    const __bf16* __restrict__ qf, const __bf16* __restrict__ kf,
    const __bf16* __restrict__ vh, float* __restrict__ merged) {
  extern __shared__ __align__(16) char smem[];
  float* St = (float*)smem;           // [DH][MP]  (S transposed)
  float* z = St + DH * MP;            // [MP]
  float* den = z + MP;                // [64]
  __bf16* Qc = (__bf16*)(den + 64);   // [64][MP]
  __bf16* Kc = Qc + 64 * MP;          // [64][MP]  natural (N-major for QK^T)
  __bf16* KcT = Kc + 64 * MP;         // [MP][64]  (N-major for V^T K)
  __bf16* Vt = KcT + MP * 64;         // [DH][64]  V transposed
  __bf16* Pm = Vt + DH * 64;          // [64][64]  masked QK^T, bf16

  const unsigned off_Qc = (unsigned)(DH * MP + MP + 64) * 4u;
  const unsigned off_Kc = off_Qc + 64u * MP * 2u;

  const int tid = threadIdx.x, lane = tid & 31, wave = tid >> 5;
  const int bh = blockIdx.x;
  const int b = bh / NH, h = bh % NH;
  const size_t fbase = (size_t)bh * TL * MP;
  const size_t vbase = (size_t)bh * TL * DH;

  for (int i = tid; i < DH * MP; i += 256) St[i] = 0.f;
  for (int i = tid; i < MP; i += 256) z[i] = 0.f;
  __syncthreads();

  for (int ch = 0; ch < TL / 64; ++ch) {
    const int t0 = ch * 64;
    // ---- stage chunk: Q and K via Tensor Data Mover (contiguous 36864B) ----
    if (wave == 0) {
      tdm_load_1d(off_Qc, qf + fbase + (size_t)t0 * MP, 64 * MP);
      tdm_load_1d(off_Kc, kf + fbase + (size_t)t0 * MP, 64 * MP);
    }
    // transposed copies (K^T for the state update, V^T for both V uses)
    for (int i = tid; i < 64 * MP; i += 256) {
      int s = i / MP, m = i % MP;
      KcT[m * 64 + s] = kf[fbase + (size_t)(t0 + s) * MP + m];
    }
    for (int i = tid; i < 64 * DH; i += 256) {
      int s = i >> 6, d = i & 63;
      Vt[d * 64 + s] = vh[vbase + (size_t)t0 * DH + i];
    }
    if (ch + 1 < TL / 64) {
      __builtin_prefetch(qf + fbase + (size_t)(t0 + 64) * MP + tid * 64, 0, 1);
      __builtin_prefetch(kf + fbase + (size_t)(t0 + 64) * MP + tid * 64, 0, 1);
    }
    if (wave == 0) __builtin_amdgcn_s_wait_tensorcnt(0);
    __syncthreads();

    // ---- A = Q K^T, strict causal mask -> Pm (bf16) ----
    for (int u = 0; u < 2; ++u) {
      int ti = wave * 2 + u, rt = ti >> 2, ct = ti & 3;
      v8f acc = {};
      if (rt >= ct) {
        for (int k0 = 0; k0 < MP; k0 += 32) {
          v16bf a = frag_a(Qc, MP, rt * 16, k0, lane);
          v16bf bb = frag_bT(Kc, MP, k0, ct * 16, lane);
          acc = wmma_bf16(a, bb, acc);
        }
      }
      const int N = lane & 15, Mo = (lane >> 4) << 3;
#pragma unroll
      for (int v = 0; v < 8; ++v) {
        int i = rt * 16 + v + Mo, j = ct * 16 + N;
        Pm[i * 64 + j] = (__bf16)((i > j) ? acc[v] : 0.f);
      }
    }
    __syncthreads();

    // ---- denominator: q.(z_prev+eps) + rowsum(Pm) ----
    if (tid < 64) {
      float d = 0.f;
      const __bf16* qrow = Qc + tid * MP;
      for (int m = 0; m < MP; ++m) d += (float)qrow[m] * (z[m] + EPSD);
      const __bf16* prow = Pm + tid * 64;
      for (int j = 0; j < 64; ++j) d += (float)prow[j];
      den[tid] = d;
    }
    __syncthreads();

    // ---- out = Q @ S_prev + Pm @ V ; /den ; store ----
    for (int u = 0; u < 2; ++u) {
      int ti = wave * 2 + u, rt = ti >> 2, ct = ti & 3;
      v8f acc = {};
      for (int k0 = 0; k0 < MP; k0 += 32) {
        v16bf a = frag_a(Qc, MP, rt * 16, k0, lane);
        v16bf bb = frag_bT_f32(St, MP, k0, ct * 16, lane);
        acc = wmma_bf16(a, bb, acc);
      }
#pragma unroll
      for (int k0 = 0; k0 < 64; k0 += 32) {
        v16bf a = frag_a(Pm, 64, rt * 16, k0, lane);
        v16bf bb = frag_bT(Vt, 64, k0, ct * 16, lane);
        acc = wmma_bf16(a, bb, acc);
      }
      const int N = lane & 15, Mo = (lane >> 4) << 3;
#pragma unroll
      for (int v = 0; v < 8; ++v) {
        int i = rt * 16 + v + Mo, dcol = ct * 16 + N;
        merged[((size_t)b * TL + (t0 + i)) * (NH * DH) + h * DH + dcol] =
            acc[v] / den[i];
      }
    }
    __syncthreads();

    // ---- state update: St += V^T K  (out rows=d, cols=m); z += sum_s k ----
    for (int u = 0; u < 9; ++u) {  // 4x18 = 72 tiles, 9 per wave
      int ti = wave * 9 + u, dt = ti / 18, mt = ti % 18;
      const int N = lane & 15, Mo = (lane >> 4) << 3;
      v8f acc;
#pragma unroll
      for (int v = 0; v < 8; ++v)
        acc[v] = St[(dt * 16 + v + Mo) * MP + mt * 16 + N];
#pragma unroll
      for (int k0 = 0; k0 < 64; k0 += 32) {
        v16bf a = frag_a(Vt, 64, dt * 16, k0, lane);    // A[d][s]
        v16bf bb = frag_bT(KcT, 64, k0, mt * 16, lane); // B[s][m], N-major
        acc = wmma_bf16(a, bb, acc);
      }
#pragma unroll
      for (int v = 0; v < 8; ++v)
        St[(dt * 16 + v + Mo) * MP + mt * 16 + N] = acc[v];
    }
    for (int m = tid; m < MP; m += 256) {
      float zz = z[m];
      const __bf16* kc = KcT + m * 64;
      for (int s = 0; s < 64; ++s) zz += (float)kc[s];
      z[m] = zz;
    }
    __syncthreads();
  }
}

// ================= kernel 4: output projection =================
__global__ __launch_bounds__(256) void outproj_kernel(
    const float* __restrict__ merged, const float* __restrict__ Wo,
    const float* __restrict__ bo, float* __restrict__ out) {
  extern __shared__ __align__(16) char smem[];
  __bf16* Ms = (__bf16*)smem;       // [64][512]
  __bf16* Ws = Ms + 64 * 512;       // [64][512] N-major (=Wo layout [e][o])
  const int tid = threadIdx.x, lane = tid & 31, wave = tid >> 5;
  const int row0 = blockIdx.x * 64;

  for (int i = tid; i < 64 * 512; i += 256)
    Ms[i] = (__bf16)merged[(size_t)row0 * 512 + i];
  for (int i = tid; i < 64 * 512; i += 256)
    Ws[i] = (__bf16)Wo[i];          // direct: Wo is [e][o]
  __syncthreads();

  for (int u = 0; u < 2; ++u) {
    int ti = wave * 2 + u, rt = ti >> 2, ct = ti & 3;
    v8f acc = {};
    for (int k0 = 0; k0 < 512; k0 += 32) {
      v16bf a = frag_a(Ms, 512, rt * 16, k0, lane);
      v16bf bb = frag_bT(Ws, 512, k0, ct * 16, lane);
      acc = wmma_bf16(a, bb, acc);
    }
    const int N = lane & 15, Mo = (lane >> 4) << 3;
#pragma unroll
    for (int v = 0; v < 8; ++v) {
      int row = row0 + rt * 16 + v + Mo;
      int e = ct * 16 + N;
      int bi = row / TL, t = row % TL;
      out[((size_t)t * BS + bi) * DH + e] = acc[v] + bo[e];
    }
  }
}

__global__ void init_kmax(unsigned* k) {
  if (threadIdx.x == 0) *k = 0u;
}

// ================= host launcher =================
extern "C" void kernel_launch(void* const* d_in, const int* in_sizes, int n_in,
                              void* d_out, int out_size, void* d_ws,
                              size_t ws_size, hipStream_t stream) {
  const float* query = (const float*)d_in[0];
  const float* key = (const float*)d_in[1];
  const float* value = (const float*)d_in[2];
  const float* Wq = (const float*)d_in[3];
  const float* bq = (const float*)d_in[4];
  const float* Wk = (const float*)d_in[5];
  const float* bk = (const float*)d_in[6];
  const float* Wv = (const float*)d_in[7];
  const float* bv = (const float*)d_in[8];
  const float* Wo = (const float*)d_in[9];
  const float* bo = (const float*)d_in[10];
  const float* proj = (const float*)d_in[11];
  float* out = (float*)d_out;

  char* ws = (char*)d_ws;
  size_t off = 0;
  auto wsalloc = [&](size_t bytes) -> void* {
    void* p = ws + off;
    off = (off + bytes + 255) & ~(size_t)255;
    return p;
  };
  unsigned* kmax = (unsigned*)wsalloc(sizeof(unsigned));
  __bf16* qh = (__bf16*)wsalloc((size_t)BHT * DH * 2);
  __bf16* kh = (__bf16*)wsalloc((size_t)BHT * DH * 2);
  __bf16* vh = (__bf16*)wsalloc((size_t)BHT * DH * 2);
  __bf16* qf = (__bf16*)wsalloc((size_t)BHT * MP * 2);
  __bf16* kf = (__bf16*)wsalloc((size_t)BHT * MP * 2);
  float* merged = (float*)wsalloc((size_t)BS * TL * NH * DH * 4);

  const size_t shm1 = (64 * 64 + 512 * 64) * 2;                       // 72 KB
  const size_t shm2 = (64 * 64 + MP * 64) * 2 + (64 * MP + 128) * 4;  // ~117 KB
  const size_t shm3 = (DH * MP + MP + 64) * 4 +
                      (64 * MP * 2 + MP * 64 + DH * 64 + 64 * 64) * 2;  // ~198 KB
  const size_t shm4 = (64 * 512 + 64 * 512) * 2;                      // 128 KB

  const int rows = TL * BS;  // 32768

  init_kmax<<<1, 64, 0, stream>>>(kmax);

  proj_kernel<<<rows / 64, 256, shm1, stream>>>(query, Wq, bq, qh, QSCALE);
  proj_kernel<<<rows / 64, 256, shm1, stream>>>(key, Wk, bk, kh, 1.0f);
  proj_kernel<<<rows / 64, 256, shm1, stream>>>(value, Wv, bv, vh, 1.0f);

  // keys: global-max reduction pass, then write pass; queries: row-max, write
  feature_kernel<<<BHT / 64, 256, shm2, stream>>>(kh, proj, kf, kmax, 0, 0);
  feature_kernel<<<BHT / 64, 256, shm2, stream>>>(kh, proj, kf, kmax, 0, 1);
  feature_kernel<<<BHT / 64, 256, shm2, stream>>>(qh, proj, qf, kmax, 1, 1);

  attn_kernel<<<BS * NH, 256, shm3, stream>>>(qf, kf, vh, merged);

  outproj_kernel<<<rows / 64, 256, shm4, stream>>>(merged, Wo, bo, out);
}
